// MoE_81655918231988
// MI455X (gfx1250) — matrix operand order repaired and checked by reading, not compiled, not run on hardware
//
#include <hip/hip_runtime.h>
#include <hip/hip_bf16.h>
#include <math.h>

// ---------------------------------------------------------------------------
// MoE forward for gfx1250 (CDNA5): bf16 WMMA expert GEMMs, f32 accumulate.
// B=8, T=512, D=512, H=1024, E=8, K=4.  NT = B*T = 4096 tokens.
// M-tile = 32 tokens/WG (8 waves): each packed B fragment feeds 2 WMMAs.
// Gate folded into hidden rows -> FFN2 accumulates via WMMA C operand.
// B-fragment loads use base-pointer + constant-IOFFSET addressing.
// ---------------------------------------------------------------------------

#define NTOK   4096
#define DDIM   512
#define HDIM   1024
#define NEXP   8
#define TOPK   4
#define LOSS_COEF 0.01f

typedef __attribute__((ext_vector_type(16))) __bf16 bf16x16;
typedef __attribute__((ext_vector_type(8)))  __bf16 bf16x8;
typedef __attribute__((ext_vector_type(8)))  float  floatx8;

#define ZERO8 ((floatx8){0.f,0.f,0.f,0.f,0.f,0.f,0.f,0.f})

// A-fragment (16x32 bf16, per-lane: row = lane&15, k-runs {0..7,16..23}+off)
#define LOAD_A(ARR, ROW, KOFF) ({                                            \
    const bf16x8 _lo = *(const bf16x8*)&ARR[ROW][KOFF];                      \
    const bf16x8 _hi = *(const bf16x8*)&ARR[ROW][(KOFF) + 16];               \
    __builtin_shufflevector(_lo, _hi, 0,1,2,3,4,5,6,7,8,9,10,11,12,13,14,15); })

#define WMMA_BF16(A, B, C) \
    __builtin_amdgcn_wmma_f32_16x16x32_bf16(false, (A), false, (B), (short)0, (C), false, false)

// ---------------------------------------------------------------------------
// 1) x (f32) -> bf16, row-major [NTOK, DDIM]
// ---------------------------------------------------------------------------
__global__ void moe_cvt_x_kernel(const float* __restrict__ x,
                                 __bf16* __restrict__ xb, int n) {
    int i = blockIdx.x * blockDim.x + threadIdx.x;
    if (i < n) xb[i] = (__bf16)x[i];
}

// ---------------------------------------------------------------------------
// 2) Pack W1 [E, D, H] (K=D rows, N=H cols, row-major) into B-fragment order:
//    per expert: frag = (kt * 64 + nt), 512 bf16 each, elem (L*16 + ee):
//      k = kt*32 + ((L & 16) >> 1) + (ee & 7) + ((ee & 8) << 1) ;  n = nt*16 + (L & 15)
// ---------------------------------------------------------------------------
__global__ void moe_pack_w1_kernel(const float* __restrict__ W1,
                                   __bf16* __restrict__ w1p) {
    int p = blockIdx.x * blockDim.x + threadIdx.x;   // 0 .. 8*512*1024-1
    int ee   = p & 15;
    int L    = (p >> 4) & 31;
    int frag = (p >> 9) & 1023;       // 16 ktiles * 64 ntiles
    int e    = p >> 19;               // 2^19 elems / expert
    int kt = frag >> 6;
    int nt = frag & 63;
    int k = kt * 32 + ((L & 16) >> 1) + (ee & 7) + ((ee & 8) << 1);
    int n = nt * 16 + (L & 15);
    w1p[p] = (__bf16)W1[(size_t)e * (DDIM * HDIM) + (size_t)k * HDIM + n];
}

// Pack W2 [E, H, D] (K=H rows, N=D cols): 32 ktiles * 32 ntiles
__global__ void moe_pack_w2_kernel(const float* __restrict__ W2,
                                   __bf16* __restrict__ w2p) {
    int p = blockIdx.x * blockDim.x + threadIdx.x;   // 0 .. 8*1024*512-1
    int ee   = p & 15;
    int L    = (p >> 4) & 31;
    int frag = (p >> 9) & 1023;       // 32 ktiles * 32 ntiles
    int e    = p >> 19;
    int kt = frag >> 5;
    int nt = frag & 31;
    int k = kt * 32 + ((L & 16) >> 1) + (ee & 7) + ((ee & 8) << 1);
    int n = nt * 16 + (L & 15);
    w2p[p] = (__bf16)W2[(size_t)e * (HDIM * DDIM) + (size_t)k * DDIM + n];
}

// ---------------------------------------------------------------------------
// 3) Gating: logits = x @ w_gate -> top-4 -> softmax -> masked dense gates
// ---------------------------------------------------------------------------
__global__ void moe_gate_kernel(const float* __restrict__ x,
                                const int* __restrict__ mask,
                                const float* __restrict__ w_gate,
                                float* __restrict__ gates) {
    int t = blockIdx.x * blockDim.x + threadIdx.x;
    if (t >= NTOK) return;
    float acc[NEXP];
    #pragma unroll
    for (int e = 0; e < NEXP; ++e) acc[e] = 0.0f;
    const float* xr = x + (size_t)t * DDIM;
    for (int d = 0; d < DDIM; ++d) {
        float xv = xr[d];
        #pragma unroll
        for (int e = 0; e < NEXP; ++e) acc[e] += xv * w_gate[d * NEXP + e];
    }
    // top-4 (keeps first/lowest index on ties, like lax.top_k)
    float tv[TOPK]; int ti[TOPK];
    #pragma unroll
    for (int j = 0; j < TOPK; ++j) {
        float best = -3.4e38f; int bi = 0;
        #pragma unroll
        for (int e = 0; e < NEXP; ++e)
            if (acc[e] > best) { best = acc[e]; bi = e; }
        tv[j] = best; ti[j] = bi; acc[bi] = -3.4e38f;
    }
    float m0 = tv[0], s = 0.0f, g[TOPK];
    #pragma unroll
    for (int j = 0; j < TOPK; ++j) { g[j] = __expf(tv[j] - m0); s += g[j]; }
    float inv = 1.0f / s;
    float mk  = (float)mask[t];
    float go[NEXP];
    #pragma unroll
    for (int e = 0; e < NEXP; ++e) go[e] = 0.0f;
    #pragma unroll
    for (int j = 0; j < TOPK; ++j) go[ti[j]] = g[j] * inv * mk;
    #pragma unroll
    for (int e = 0; e < NEXP; ++e) gates[(size_t)t * NEXP + e] = go[e];
}

// ---------------------------------------------------------------------------
// 4) Load-balancing loss: importance = sum_t gates -> var(ddof=1)/mean^2 * c
// ---------------------------------------------------------------------------
__global__ void moe_loss_kernel(const float* __restrict__ gates,
                                float* __restrict__ loss_out) {
    __shared__ float red[NEXP][256];
    int tid = threadIdx.x;
    float s[NEXP];
    #pragma unroll
    for (int e = 0; e < NEXP; ++e) s[e] = 0.0f;
    for (int t = tid; t < NTOK; t += 256) {
        #pragma unroll
        for (int e = 0; e < NEXP; ++e) s[e] += gates[(size_t)t * NEXP + e];
    }
    #pragma unroll
    for (int e = 0; e < NEXP; ++e) red[e][tid] = s[e];
    __syncthreads();
    for (int off = 128; off > 0; off >>= 1) {
        if (tid < off) {
            #pragma unroll
            for (int e = 0; e < NEXP; ++e) red[e][tid] += red[e][tid + off];
        }
        __syncthreads();
    }
    if (tid == 0) {
        float mean = 0.0f;
        #pragma unroll
        for (int e = 0; e < NEXP; ++e) mean += red[e][0];
        mean *= (1.0f / NEXP);
        float var = 0.0f;
        #pragma unroll
        for (int e = 0; e < NEXP; ++e) {
            float d = red[e][0] - mean;
            var += d * d;
        }
        var *= (1.0f / (NEXP - 1));                  // unbiased (ddof=1)
        loss_out[0] = (var / (mean * mean + 1e-10f)) * LOSS_COEF;
    }
}

// ---------------------------------------------------------------------------
// 5) Expert FFN: WG = 32 tokens, 256 threads (8 waves), H in 4 chunks of 256.
//    h = gate .* relu(X@W1_e + b1_e)  (bf16 in LDS, gate folded in)
//    out (+)= h@W2_e  via WMMA C-accumulation;  + gate.*b2_e once per expert
//    Epilogue: y = sigmoid(out) + x
// ---------------------------------------------------------------------------
__global__ __launch_bounds__(256) void moe_ffn_kernel(
    const __bf16* __restrict__ xb,   const __bf16* __restrict__ w1p,
    const __bf16* __restrict__ w2p,  const float*  __restrict__ b1,
    const float*  __restrict__ b2,   const float*  __restrict__ gates,
    const float*  __restrict__ xf,   float* __restrict__ out) {

    // padded rows (stride % 64 banks == +4) -> conflict-light ds_load_b128
    __shared__ __align__(16) __bf16 Xl[32][520];   // 33,280 B
    __shared__ __align__(16) __bf16 Hl[32][264];   // 16,896 B (one 256-col chunk)
    __shared__ float Gl[32][NEXP];                 //  1,024 B

    const int tid  = threadIdx.x;
    const int lane = tid & 31;
    const int wv   = tid >> 5;            // wave 0..7
    const int tok0 = blockIdx.x * 32;
    const int mrow = lane & 15;           // A-matrix row held by this lane
    const int off8 = (lane & 16) >> 1;    // 0 (lanes 0-15) / 8 (lanes 16-31)
    const int ncol = lane & 15;           // C/D column held by this lane

    // ---- stage X tile (32 x 512 bf16, 16B vector copies) + gates (32x8) ----
    for (int i = tid; i < 32 * (DDIM / 8); i += 256) {
        int m = i >> 6;                  // 64 x 16B chunks per row
        int c = (i & 63) * 8;
        *(bf16x8*)&Xl[m][c] = *(const bf16x8*)&xb[(size_t)(tok0 + m) * DDIM + c];
    }
    Gl[tid >> 3][tid & 7] = gates[(size_t)(tok0 + (tid >> 3)) * NEXP + (tid & 7)];
    __syncthreads();

    // persistent output accumulators: [mt][j] -> rows mt*16.., cols (wv*4+j)*16..
    floatx8 oacc[2][4];
    #pragma unroll
    for (int mt = 0; mt < 2; ++mt)
        #pragma unroll
        for (int j = 0; j < 4; ++j) oacc[mt][j] = ZERO8;

    for (int e = 0; e < NEXP; ++e) {
        // per-lane gate rows: gta[mt][r] = gate(token mt*16 + r + off8)
        float gta[2][8]; float gs = 0.0f;
        #pragma unroll
        for (int mt = 0; mt < 2; ++mt)
            #pragma unroll
            for (int r = 0; r < 8; ++r) {
                gta[mt][r] = Gl[mt * 16 + r + off8][e];
                gs += gta[mt][r];
            }
        if (__ballot(gs != 0.0f) == 0ull) continue;   // uniform across WG

        const __bf16* w1e = w1p + (size_t)e * (DDIM * HDIM);
        const __bf16* w2e = w2p + (size_t)e * (HDIM * DDIM);

        // gate .* b2 contribution (once per expert)
        #pragma unroll
        for (int j = 0; j < 4; ++j) {
            const float b2v = b2[e * DDIM + (wv * 4 + j) * 16 + ncol];
            #pragma unroll
            for (int mt = 0; mt < 2; ++mt)
                #pragma unroll
                for (int r = 0; r < 8; ++r) oacc[mt][j][r] += gta[mt][r] * b2v;
        }

        for (int c = 0; c < 4; ++c) {       // H chunks of 256 columns
            // ---- FFN1: wave owns nt_global = c*16 + wv*2 + {0,1} ----
            // preload biases early so their wait lands after the GEMM
            const float bias0 = b1[e * HDIM + (c * 16 + wv * 2) * 16 + ncol];
            const float bias1 = b1[e * HDIM + (c * 16 + wv * 2 + 1) * 16 + ncol];

            floatx8 acc[2][2];              // [mt][j]
            #pragma unroll
            for (int mt = 0; mt < 2; ++mt)
                #pragma unroll
                for (int j = 0; j < 2; ++j) acc[mt][j] = ZERO8;

            // base ptr: frag (kt=0, nt = c*16 + wv*2), lane slice; kt stride
            // = 64 frags * 512 elems = 32768 elems (65536 B); j stride = 512
            const __bf16* pw1 =
                w1e + (size_t)(((c * 16 + wv * 2) * 32 + lane) * 16);
            #pragma unroll 4
            for (int kt = 0; kt < 16; ++kt) {
                const bf16x16 a0 = LOAD_A(Xl, mrow,      kt * 32 + off8);
                const bf16x16 a1 = LOAD_A(Xl, 16 + mrow, kt * 32 + off8);
                const bf16x16 b0 = *(const bf16x16*)(pw1);
                const bf16x16 b1f = *(const bf16x16*)(pw1 + 512);
                __builtin_prefetch(pw1 + 2 * 32768, 0, 1);
                acc[0][0] = WMMA_BF16(a0, b0,  acc[0][0]);
                acc[1][0] = WMMA_BF16(a1, b0,  acc[1][0]);
                acc[0][1] = WMMA_BF16(a0, b1f, acc[0][1]);
                acc[1][1] = WMMA_BF16(a1, b1f, acc[1][1]);
                pw1 += 32768;
            }
            // relu + bias, fold gate, store bf16 hidden chunk
            #pragma unroll
            for (int j = 0; j < 2; ++j) {
                const int ncl  = (wv * 2 + j) * 16 + ncol;   // chunk-local col
                const float bias = j ? bias1 : bias0;
                #pragma unroll
                for (int mt = 0; mt < 2; ++mt)
                    #pragma unroll
                    for (int r = 0; r < 8; ++r) {
                        float v = acc[mt][j][r] + bias;
                        v = v > 0.0f ? v : 0.0f;             // ReLU
                        v *= gta[mt][r];                     // fold gate
                        Hl[mt * 16 + r + off8][ncl] = (__bf16)v;
                    }
            }
            __syncthreads();

            // ---- FFN2 partial: k in [c*256, c*256+256) -> accumulate oacc ----
            // base ptr: frag (kt = c*8, nt2 = wv*4); ktl stride = 32 frags *
            // 512 elems = 16384 elems (32768 B); j stride = 512 elems
            const __bf16* pw2 =
                w2e + (size_t)(((c * 8 * 32 + wv * 4) * 32 + lane) * 16);
            #pragma unroll 2
            for (int ktl = 0; ktl < 8; ++ktl) {
                const bf16x16 a0 = LOAD_A(Hl, mrow,      ktl * 32 + off8);
                const bf16x16 a1 = LOAD_A(Hl, 16 + mrow, ktl * 32 + off8);
                __builtin_prefetch(pw2 + 2 * 16384, 0, 1);
                #pragma unroll
                for (int j = 0; j < 4; ++j) {
                    const bf16x16 bfr = *(const bf16x16*)(pw2 + j * 512);
                    oacc[0][j] = WMMA_BF16(a0, bfr, oacc[0][j]);
                    oacc[1][j] = WMMA_BF16(a1, bfr, oacc[1][j]);
                }
                pw2 += 16384;
            }
            __syncthreads();   // before next chunk overwrites Hl
        }
    }

    // ---- epilogue: y = sigmoid(combined) + x ----
    #pragma unroll
    for (int j = 0; j < 4; ++j) {
        const int n = (wv * 4 + j) * 16 + ncol;
        #pragma unroll
        for (int mt = 0; mt < 2; ++mt)
            #pragma unroll
            for (int r = 0; r < 8; ++r) {
                const int m = mt * 16 + r + off8;
                const size_t idx = (size_t)(tok0 + m) * DDIM + n;
                const float v = oacc[mt][j][r];
                out[idx] = 1.0f / (1.0f + __expf(-v)) + xf[idx];
            }
    }
}

// ---------------------------------------------------------------------------
// Launch
// ---------------------------------------------------------------------------
extern "C" void kernel_launch(void* const* d_in, const int* in_sizes, int n_in,
                              void* d_out, int out_size, void* d_ws, size_t ws_size,
                              hipStream_t stream) {
    const float* x      = (const float*)d_in[0];
    const int*   mask   = (const int*)d_in[1];
    const float* w_gate = (const float*)d_in[2];
    const float* W1     = (const float*)d_in[3];
    const float* b1     = (const float*)d_in[4];
    const float* W2     = (const float*)d_in[5];
    const float* b2     = (const float*)d_in[6];
    float* out = (float*)d_out;

    // workspace layout
    char* ws = (char*)d_ws;
    float*  gates = (float*)(ws);                                   // 128 KB
    __bf16* xb    = (__bf16*)(ws + 131072);                         // 4 MB
    __bf16* w1p   = (__bf16*)(ws + 131072 + 4194304);               // 8 MB
    __bf16* w2p   = (__bf16*)(ws + 131072 + 4194304 + 8388608);     // 8 MB

    const int nx = NTOK * DDIM;                 // 2,097,152
    const int nw = NEXP * DDIM * HDIM;          // 4,194,304

    moe_cvt_x_kernel<<<(nx + 255) / 256, 256, 0, stream>>>(x, xb, nx);
    moe_pack_w1_kernel<<<nw / 256, 256, 0, stream>>>(W1, w1p);
    moe_pack_w2_kernel<<<nw / 256, 256, 0, stream>>>(W2, w2p);
    moe_gate_kernel<<<NTOK / 256, 256, 0, stream>>>(x, mask, w_gate, gates);
    moe_loss_kernel<<<1, 256, 0, stream>>>(gates, out + (out_size - 1));
    moe_ffn_kernel<<<NTOK / 32, 256, 0, stream>>>(xb, w1p, w2p, b1, b2,
                                                  gates, x, out);
}
